// TransLayer_27925877358772
// MI455X (gfx1250) — compile-verified, matrix operand order checked by reading, not run on hardware
//
#include <hip/hip_runtime.h>
#include <hip/hip_bf16.h>
#include <math.h>

// ---------------------------------------------------------------------------
// CDNA5 / gfx1250 implementation: all GEMMs through v_wmma_f32_16x16x32_bf16.
// Double-buffered LDS pipeline; A-tile staging uses GLOBAL_LOAD_ASYNC_TO_LDS
// (ASYNCcnt) when the toolchain exposes the builtin, else a register path.
// ---------------------------------------------------------------------------

typedef __attribute__((ext_vector_type(16))) __bf16 v16bf;
typedef __attribute__((ext_vector_type(8)))  float  v8f;
typedef __attribute__((ext_vector_type(4)))  int    v4i;
typedef unsigned short ushort_t;

#define GAS __attribute__((address_space(1)))
#define LAS __attribute__((address_space(3)))

#if defined(__has_builtin)
#  if __has_builtin(__builtin_amdgcn_global_load_async_to_lds_b128)
#    define HAVE_ASYNC_LDS 1
#  endif
#endif
#ifndef HAVE_ASYNC_LDS
#  define HAVE_ASYNC_LDS 0
#endif

// copy 16B global -> LDS; async DMA on CDNA5 (no VGPR round trip)
__device__ __forceinline__ void copy16_to_lds(const ushort_t* g, ushort_t* l) {
#if HAVE_ASYNC_LDS
    v4i* gp = (v4i*)(void*)const_cast<ushort_t*>(g);   // generic typed ptrs
    v4i* lp = (v4i*)(void*)l;
    __builtin_amdgcn_global_load_async_to_lds_b128(
        (GAS v4i*)gp, (LAS v4i*)lp, 0, 0);
#else
    *(uint4*)l = *(const uint4*)g;
#endif
}

__device__ __forceinline__ void async_wait() {
#if HAVE_ASYNC_LDS
#  if __has_builtin(__builtin_amdgcn_s_wait_asynccnt)
    __builtin_amdgcn_s_wait_asynccnt(0);
#  else
    asm volatile("s_wait_asynccnt 0x0" ::: "memory");
#  endif
#endif
}

__device__ __forceinline__ ushort_t f2bf(float f) {
    unsigned int u = __float_as_uint(f);
    u += 0x7FFFu + ((u >> 16) & 1u);      // round-to-nearest-even
    return (ushort_t)(u >> 16);
}

// problem constants
#define NROWS 2048
#define DBv   4096
#define DAv   5120
#define DX2v  1024
#define MDL   512

// GEMM tiling: 256x64 block tile, 8 wave32 waves (4 along M x 2 along N),
// each wave 64x32 = 4x2 WMMA 16x16 fragments, K step 32 (bf16 WMMA K).
#define BM 256
#define BN 64
#define BK 32
#define SA_STRIDE 40   // 32 + 8 pad (keeps 16B alignment, dodges bank conflicts)
#define SB_STRIDE 40

union FragBF { uint4 q[2]; v16bf v; };
union FragF  { float f[8];  v8f  v; };

// fragment loads (all up-front, so ds waits stagger) + 8 WMMAs for one K-tile
__device__ __forceinline__ void wmma_tile(const ushort_t* bufA, const ushort_t* bufB,
                                          FragF (&acc)[4][2], int aBase, int bBase)
{
    FragBF bfr[2], afr[4];
    #pragma unroll
    for (int nt = 0; nt < 2; ++nt) {
        const ushort_t* p = bufB + bBase + nt * (16 * SB_STRIDE);
        bfr[nt].q[0] = *(const uint4*)p;
        bfr[nt].q[1] = *(const uint4*)(p + 16);
    }
    #pragma unroll
    for (int mt = 0; mt < 4; ++mt) {
        const ushort_t* p = bufA + aBase + mt * (16 * SA_STRIDE);
        afr[mt].q[0] = *(const uint4*)p;
        afr[mt].q[1] = *(const uint4*)(p + 16);
    }
    #pragma unroll
    for (int mt = 0; mt < 4; ++mt)
        #pragma unroll
        for (int nt = 0; nt < 2; ++nt)
            acc[mt][nt].v = __builtin_amdgcn_wmma_f32_16x16x32_bf16(
                false, afr[mt].v, false, bfr[nt].v,
                (short)0, acc[mt][nt].v, false, false);
}

// C = alpha * (A @ B) + bias, optional relu; A row-major bf16 [M,K] (ldA),
// B row-major bf16 [K,N] (ldB) or, if TRANSB, row-major [N,K] (ldB);
// outputs: f32 (Cf) and/or bf16 (Ch), both with ldC; blockIdx.z batching via
// element strides strideA/strideB/strideC.
template<int TRANSB>
__global__ __launch_bounds__(256) void gemm_bf16_wmma(
    const ushort_t* __restrict__ A, long long strideA, int ldA,
    const ushort_t* __restrict__ B, long long strideB, int ldB,
    float* Cf, ushort_t* Ch, long long strideC, int ldC,
    const float* __restrict__ bias,
    int M, int N, int K, float alpha, int relu)
{
    __shared__ ushort_t sA[2][BM * SA_STRIDE];
    __shared__ ushort_t sB[2][BN * SB_STRIDE];   // stored transposed: [n][k]

    const int tid   = threadIdx.x;
    const int lane  = tid & 31;
    const int wid   = tid >> 5;
    const int waveM = wid >> 1;    // 0..3 -> 64-row stripes
    const int waveN = wid & 1;     // 0..1 -> 32-col stripes
    const int half  = lane >> 4;   // K-half selector in fragment layout
    const int l16   = lane & 15;

    const int nBase = blockIdx.x * BN;
    const int mBase = blockIdx.y * BM;
    const int z     = blockIdx.z;

    A += (long long)z * strideA;
    B += (long long)z * strideB;
    const long long coff = (long long)z * strideC;

    // staging indices (A: 1024 16B chunks; B trans: 256 16B chunks; B: 512 8B chunks)
    const int aRow[4] = { (tid + 0)   >> 2, (tid + 256) >> 2,
                          (tid + 512) >> 2, (tid + 768) >> 2 };
    const int aCol    = (tid & 3) << 3;
    const int btRow   = tid >> 2;            // TRANSB path
    const int btCol   = (tid & 3) << 3;
    const int bK0     = (tid + 0)   >> 4;    // scatter path
    const int bK1     = (tid + 256) >> 4;
    const int bN0     = ((tid + 0)   & 15) << 2;
    const int bN1     = ((tid + 256) & 15) << 2;

    // fragment LDS base offsets (loop-invariant)
    const int aBase = (waveM * 64 + l16) * SA_STRIDE + half * 8;
    const int bBase = (waveN * 32 + l16) * SB_STRIDE + half * 8;

    FragF acc[4][2];
    #pragma unroll
    for (int i = 0; i < 4; ++i)
        #pragma unroll
        for (int j = 0; j < 2; ++j)
            #pragma unroll
            for (int r = 0; r < 8; ++r) acc[i][j].f[r] = 0.f;

    const int nTiles = K / BK;   // always >= 2 for this pipeline

    // ---- prologue: stage tile 0 into buffer 0 ----
    {
        #pragma unroll
        for (int i = 0; i < 4; ++i)
            copy16_to_lds(A + (size_t)(mBase + aRow[i]) * ldA + aCol,
                          &sA[0][aRow[i] * SA_STRIDE + aCol]);
        if constexpr (TRANSB) {
            copy16_to_lds(B + (size_t)(nBase + btRow) * ldB + btCol,
                          &sB[0][btRow * SB_STRIDE + btCol]);
        } else {
            const uint2 b0 = *(const uint2*)(B + (size_t)bK0 * ldB + nBase + bN0);
            const uint2 b1 = *(const uint2*)(B + (size_t)bK1 * ldB + nBase + bN1);
            sB[0][(bN0 + 0) * SB_STRIDE + bK0] = (ushort_t)(b0.x & 0xFFFFu);
            sB[0][(bN0 + 1) * SB_STRIDE + bK0] = (ushort_t)(b0.x >> 16);
            sB[0][(bN0 + 2) * SB_STRIDE + bK0] = (ushort_t)(b0.y & 0xFFFFu);
            sB[0][(bN0 + 3) * SB_STRIDE + bK0] = (ushort_t)(b0.y >> 16);
            sB[0][(bN1 + 0) * SB_STRIDE + bK1] = (ushort_t)(b1.x & 0xFFFFu);
            sB[0][(bN1 + 1) * SB_STRIDE + bK1] = (ushort_t)(b1.x >> 16);
            sB[0][(bN1 + 2) * SB_STRIDE + bK1] = (ushort_t)(b1.y & 0xFFFFu);
            sB[0][(bN1 + 3) * SB_STRIDE + bK1] = (ushort_t)(b1.y >> 16);
        }
        async_wait();                          // tile 0 resident before first use
    }

    // ---- steady state: stage tile t+1 (async, into nxt) while computing t ----
    #pragma unroll 2
    for (int t = 0; t < nTiles - 1; ++t) {
        __syncthreads();                       // tile t visible; nxt's readers done
        const int cur = t & 1;
        const int nxt = cur ^ 1;
        const int kN  = (t + 1) * BK;

        // async DMA of A tile t+1 straight into LDS (no VGPR round trip)
        #pragma unroll
        for (int i = 0; i < 4; ++i)
            copy16_to_lds(A + (size_t)(mBase + aRow[i]) * ldA + kN + aCol,
                          &sA[nxt][aRow[i] * SA_STRIDE + aCol]);

        uint2 b0, b1;
        if constexpr (TRANSB) {
            copy16_to_lds(B + (size_t)(nBase + btRow) * ldB + kN + btCol,
                          &sB[nxt][btRow * SB_STRIDE + btCol]);
        } else {
            b0 = *(const uint2*)(B + (size_t)(kN + bK0) * ldB + nBase + bN0);
            b1 = *(const uint2*)(B + (size_t)(kN + bK1) * ldB + nBase + bN1);
        }

        wmma_tile(&sA[cur][0], &sB[cur][0], acc, aBase, bBase);

        if constexpr (!TRANSB) {               // scatter-transpose B commit
            sB[nxt][(bN0 + 0) * SB_STRIDE + bK0] = (ushort_t)(b0.x & 0xFFFFu);
            sB[nxt][(bN0 + 1) * SB_STRIDE + bK0] = (ushort_t)(b0.x >> 16);
            sB[nxt][(bN0 + 2) * SB_STRIDE + bK0] = (ushort_t)(b0.y & 0xFFFFu);
            sB[nxt][(bN0 + 3) * SB_STRIDE + bK0] = (ushort_t)(b0.y >> 16);
            sB[nxt][(bN1 + 0) * SB_STRIDE + bK1] = (ushort_t)(b1.x & 0xFFFFu);
            sB[nxt][(bN1 + 1) * SB_STRIDE + bK1] = (ushort_t)(b1.x >> 16);
            sB[nxt][(bN1 + 2) * SB_STRIDE + bK1] = (ushort_t)(b1.y & 0xFFFFu);
            sB[nxt][(bN1 + 3) * SB_STRIDE + bK1] = (ushort_t)(b1.y >> 16);
        }
        async_wait();                          // tile t+1 in LDS before next barrier
    }

    // ---- peeled last tile: no staging ----
    __syncthreads();
    {
        const int cur = (nTiles - 1) & 1;
        wmma_tile(&sA[cur][0], &sB[cur][0], acc, aBase, bBase);
    }

    // ---- epilogue: C/D layout -> lane l16 = N, VGPR r + half*8 = M ----
    #pragma unroll
    for (int mt = 0; mt < 4; ++mt) {
        #pragma unroll
        for (int nt = 0; nt < 2; ++nt) {
            const int col = nBase + waveN * 32 + nt * 16 + l16;
            const float bb = bias ? bias[col] : 0.f;
            #pragma unroll
            for (int r = 0; r < 8; ++r) {
                float v = alpha * acc[mt][nt].f[r] + bb;
                if (relu) v = fmaxf(v, 0.f);
                const int row = mBase + waveM * 64 + mt * 16 + half * 8 + r;
                const size_t idx = (size_t)coff + (size_t)row * ldC + col;
                if (Cf) Cf[idx] = v;
                if (Ch) Ch[idx] = f2bf(v);
            }
        }
    }
}

// ---------------------------------------------------------------------------
// elementwise conversion kernels
// ---------------------------------------------------------------------------
__global__ void cvt_flat(const float* __restrict__ src, ushort_t* __restrict__ dst,
                         size_t n, int sigm)
{
    size_t i = (size_t)blockIdx.x * blockDim.x + threadIdx.x;
    const size_t stride = (size_t)gridDim.x * blockDim.x;
    for (; i < n; i += stride) {
        float v = src[i];
        if (sigm) v = 1.f / (1.f + __expf(-v));
        dst[i] = f2bf(v);
    }
}

// copy f32 [rows,cols] (ldSrc) -> bf16 dst at column offset colOff (ldDst)
__global__ void cvt_cols(const float* __restrict__ src, int ldSrc,
                         ushort_t* __restrict__ dst, int ldDst, int colOff,
                         int rows, int cols)
{
    size_t i = (size_t)blockIdx.x * blockDim.x + threadIdx.x;
    const size_t stride = (size_t)gridDim.x * blockDim.x;
    const size_t n = (size_t)rows * cols;
    for (; i < n; i += stride) {
        const int r = (int)(i / cols);
        const int c = (int)(i % cols);
        dst[(size_t)r * ldDst + colOff + c] = f2bf(src[(size_t)r * ldSrc + c]);
    }
}

// softmax over the ROW axis (per column) -> bf16; thread j owns column j,
// row-loop accesses are coalesced across threads.
__global__ void col_softmax(const float* __restrict__ S, ushort_t* __restrict__ out,
                            int rows, int cols)
{
    const int j = blockIdx.x * blockDim.x + threadIdx.x;
    if (j >= cols) return;
    float m = -3.4e38f, s = 0.f;
    for (int i = 0; i < rows; ++i) {           // online max + sum
        const float x = S[(size_t)i * cols + j];
        const float nm = fmaxf(m, x);
        s = s * __expf(m - nm) + __expf(x - nm);
        m = nm;
    }
    const float inv = 1.f / s;
    for (int i = 0; i < rows; ++i)
        out[(size_t)i * cols + j] = f2bf(__expf(S[(size_t)i * cols + j] - m) * inv);
}

// standard row softmax -> bf16 (block per row)
__global__ __launch_bounds__(256) void row_softmax(const float* __restrict__ S,
                                                   ushort_t* __restrict__ out, int cols)
{
    __shared__ float red[256];
    const int row = blockIdx.x;
    const int tid = threadIdx.x;
    const float* p = S + (size_t)row * cols;
    float m = -3.4e38f;
    for (int c = tid; c < cols; c += 256) m = fmaxf(m, p[c]);
    red[tid] = m; __syncthreads();
    for (int s = 128; s > 0; s >>= 1) { if (tid < s) red[tid] = fmaxf(red[tid], red[tid + s]); __syncthreads(); }
    m = red[0]; __syncthreads();
    float sum = 0.f;
    for (int c = tid; c < cols; c += 256) sum += __expf(p[c] - m);
    red[tid] = sum; __syncthreads();
    for (int s = 128; s > 0; s >>= 1) { if (tid < s) red[tid] += red[tid + s]; __syncthreads(); }
    const float inv = 1.f / red[0];
    ushort_t* o = out + (size_t)row * cols;
    for (int c = tid; c < cols; c += 256) o[c] = f2bf(__expf(p[c] - m) * inv);
}

// out = LayerNorm(x + res) * g + b ; writes f32 and optional bf16 (block per row)
__global__ __launch_bounds__(256) void layernorm_res(
    const float* __restrict__ x, const float* __restrict__ res,
    const float* __restrict__ g, const float* __restrict__ b,
    float* __restrict__ outF, ushort_t* __restrict__ outH, int cols)
{
    __shared__ float red[256];
    __shared__ float rowv[MDL];
    const int row = blockIdx.x;
    const int tid = threadIdx.x;
    const float* px = x + (size_t)row * cols;
    const float* pr = res + (size_t)row * cols;
    float ls = 0.f;
    for (int c = tid; c < cols; c += 256) { const float v = px[c] + pr[c]; rowv[c] = v; ls += v; }
    red[tid] = ls; __syncthreads();
    for (int s = 128; s > 0; s >>= 1) { if (tid < s) red[tid] += red[tid + s]; __syncthreads(); }
    const float mean = red[0] / cols;
    __syncthreads();
    float lv = 0.f;
    for (int c = tid; c < cols; c += 256) { const float d = rowv[c] - mean; lv += d * d; }
    red[tid] = lv; __syncthreads();
    for (int s = 128; s > 0; s >>= 1) { if (tid < s) red[tid] += red[tid + s]; __syncthreads(); }
    const float inv = rsqrtf(red[0] / cols + 1e-5f);
    for (int c = tid; c < cols; c += 256) {
        const float o = (rowv[c] - mean) * inv * g[c] + b[c];
        outF[(size_t)row * cols + c] = o;
        if (outH) outH[(size_t)row * cols + c] = f2bf(o);
    }
}

// ---------------------------------------------------------------------------
// host orchestration
// ---------------------------------------------------------------------------
extern "C" void kernel_launch(void* const* d_in, const int* in_sizes, int n_in,
                              void* d_out, int out_size, void* d_ws, size_t ws_size,
                              hipStream_t stream)
{
    (void)in_sizes; (void)n_in; (void)out_size; (void)ws_size;

    const float* X1    = (const float*)d_in[0];
    const float* X2    = (const float*)d_in[1];
    const float* X3    = (const float*)d_in[2];
    const float* WQ    = (const float*)d_in[3];
    const float* WK    = (const float*)d_in[4];
    const float* WV    = (const float*)d_in[5];
    const float* WVB   = (const float*)d_in[6];
    const float* W_ali = (const float*)d_in[7];
    const float* W_O   = (const float*)d_in[8];
    const float* eWq   = (const float*)d_in[9];
    const float* eWk   = (const float*)d_in[10];
    const float* eWv   = (const float*)d_in[11];
    const float* ebq   = (const float*)d_in[12];
    const float* ebk   = (const float*)d_in[13];
    const float* ebv   = (const float*)d_in[14];
    const float* eWo   = (const float*)d_in[15];
    const float* ebo   = (const float*)d_in[16];
    const float* g1p   = (const float*)d_in[17];
    const float* b1p   = (const float*)d_in[18];
    const float* W1p   = (const float*)d_in[19];
    const float* bW1p  = (const float*)d_in[20];
    const float* W2p   = (const float*)d_in[21];
    const float* bW2p  = (const float*)d_in[22];
    const float* g2p   = (const float*)d_in[23];
    const float* b2p   = (const float*)d_in[24];

    char* ws = (char*)d_ws;
    size_t off = 0;
    auto alloc = [&](size_t bytes) -> void* {
        void* p = ws + off;
        off = (off + bytes + 255) & ~(size_t)255;
        return p;
    };

    // bf16 operand buffers
    ushort_t* XAbf   = (ushort_t*)alloc((size_t)NROWS * DAv * 2);
    ushort_t* sX3bf  = (ushort_t*)alloc((size_t)NROWS * DBv * 2);
    ushort_t* X2bf   = (ushort_t*)alloc((size_t)NROWS * DX2v * 2);
    ushort_t* wQbf   = (ushort_t*)alloc((size_t)8 * DAv * 64 * 2);
    ushort_t* wKbf   = (ushort_t*)alloc((size_t)8 * DAv * 64 * 2);
    ushort_t* wVbf   = (ushort_t*)alloc((size_t)8 * DAv * 64 * 2);
    ushort_t* wVBbf  = (ushort_t*)alloc((size_t)4 * DX2v * 64 * 2);
    ushort_t* wAlibf = (ushort_t*)alloc((size_t)DBv * NROWS * 2);
    ushort_t* wObf   = (ushort_t*)alloc((size_t)768 * MDL * 2);
    ushort_t* eWqbf  = (ushort_t*)alloc((size_t)2 * MDL * MDL * 2);
    ushort_t* eWkbf  = (ushort_t*)alloc((size_t)2 * MDL * MDL * 2);
    ushort_t* eWvbf  = (ushort_t*)alloc((size_t)2 * MDL * MDL * 2);
    ushort_t* eWobf  = (ushort_t*)alloc((size_t)2 * MDL * MDL * 2);
    ushort_t* eW1bf  = (ushort_t*)alloc((size_t)2 * MDL * 64 * 2);
    ushort_t* eW2bf  = (ushort_t*)alloc((size_t)2 * 64 * MDL * 2);
    // activations
    ushort_t* Qb     = (ushort_t*)alloc((size_t)8 * NROWS * 64 * 2);
    ushort_t* Kb     = (ushort_t*)alloc((size_t)8 * NROWS * 64 * 2);
    ushort_t* Vb     = (ushort_t*)alloc((size_t)8 * NROWS * 64 * 2);
    ushort_t* zbbf   = (ushort_t*)alloc((size_t)4 * NROWS * 64 * 2);
    ushort_t* alibf  = (ushort_t*)alloc((size_t)NROWS * NROWS * 2);
    float*    scoresF= (float*)   alloc((size_t)NROWS * NROWS * 4);
    ushort_t* Sbf    = (ushort_t*)alloc((size_t)NROWS * NROWS * 2);
    float*    ZcatF  = (float*)   alloc((size_t)NROWS * 768 * 4);
    ushort_t* Zcatbf = (ushort_t*)alloc((size_t)NROWS * 768 * 2);
    float*    xF     = (float*)   alloc((size_t)NROWS * MDL * 4);
    ushort_t* xbf    = (ushort_t*)alloc((size_t)NROWS * MDL * 2);
    ushort_t* qbf    = (ushort_t*)alloc((size_t)NROWS * MDL * 2);
    ushort_t* kbf    = (ushort_t*)alloc((size_t)NROWS * MDL * 2);
    ushort_t* vbf    = (ushort_t*)alloc((size_t)NROWS * MDL * 2);
    ushort_t* attbf  = (ushort_t*)alloc((size_t)NROWS * MDL * 2);
    float*    projF  = (float*)   alloc((size_t)NROWS * MDL * 4);
    float*    x1F    = (float*)   alloc((size_t)NROWS * MDL * 4);
    ushort_t* x1bf   = (ushort_t*)alloc((size_t)NROWS * MDL * 2);
    ushort_t* h1bf   = (ushort_t*)alloc((size_t)NROWS * 64 * 2);
    float*    fF     = (float*)   alloc((size_t)NROWS * MDL * 4);

    auto cvt = [&](const float* s, ushort_t* d, size_t n, int sig) {
        int blocks = (int)((n + 255) / 256); if (blocks > 8192) blocks = 8192;
        cvt_flat<<<dim3(blocks), dim3(256), 0, stream>>>(s, d, n, sig);
    };
    auto gemm = [&](const ushort_t* A, long long sA, int ldA,
                    const ushort_t* B, long long sB, int ldB, int tB,
                    float* Cf, ushort_t* Ch, long long sC, int ldC,
                    const float* bias, int M, int N, int K,
                    float alpha, int relu, int batch) {
        dim3 grid(N / BN, M / BM, batch);
        if (tB)
            gemm_bf16_wmma<1><<<grid, dim3(256), 0, stream>>>(
                A, sA, ldA, B, sB, ldB, Cf, Ch, sC, ldC, bias, M, N, K, alpha, relu);
        else
            gemm_bf16_wmma<0><<<grid, dim3(256), 0, stream>>>(
                A, sA, ldA, B, sB, ldB, Cf, Ch, sC, ldC, bias, M, N, K, alpha, relu);
    };

    // ---- operand conversion ----
    {
        int blk = 8192;
        cvt_cols<<<dim3(blk), dim3(256), 0, stream>>>(X1, DBv,  XAbf, DAv, 0,    NROWS, DBv);
        cvt_cols<<<dim3(blk), dim3(256), 0, stream>>>(X2, DX2v, XAbf, DAv, DBv,  NROWS, DX2v);
    }
    cvt(X2,    X2bf,   (size_t)NROWS * DX2v, 0);
    cvt(X3,    sX3bf,  (size_t)NROWS * DBv,  1);      // fused sigmoid
    cvt(WQ,    wQbf,   (size_t)8 * DAv * 64, 0);
    cvt(WK,    wKbf,   (size_t)8 * DAv * 64, 0);
    cvt(WV,    wVbf,   (size_t)8 * DAv * 64, 0);
    cvt(WVB,   wVBbf,  (size_t)4 * DX2v * 64, 0);
    cvt(W_ali, wAlibf, (size_t)DBv * NROWS, 0);
    cvt(W_O,   wObf,   (size_t)768 * MDL, 0);
    cvt(eWq,   eWqbf,  (size_t)2 * MDL * MDL, 0);
    cvt(eWk,   eWkbf,  (size_t)2 * MDL * MDL, 0);
    cvt(eWv,   eWvbf,  (size_t)2 * MDL * MDL, 0);
    cvt(eWo,   eWobf,  (size_t)2 * MDL * MDL, 0);
    cvt(W1p,   eW1bf,  (size_t)2 * MDL * 64, 0);
    cvt(W2p,   eW2bf,  (size_t)2 * 64 * MDL, 0);

    const long long hW  = (long long)DAv * 64;     // per-head WQ/WK/WV stride
    const long long hQ  = (long long)NROWS * 64;   // per-head Q/K/V stride

    // ---- type-A heads: Q/K/V projections (batched over 8 heads) ----
    gemm(XAbf, 0, DAv, wQbf, hW, 64, 0, nullptr, Qb, hQ, 64, nullptr, NROWS, 64, DAv, 1.f, 0, 8);
    gemm(XAbf, 0, DAv, wKbf, hW, 64, 0, nullptr, Kb, hQ, 64, nullptr, NROWS, 64, DAv, 1.f, 0, 8);
    gemm(XAbf, 0, DAv, wVbf, hW, 64, 0, nullptr, Vb, hQ, 64, nullptr, NROWS, 64, DAv, 1.f, 0, 8);

    // ---- type-B: zb = X2 @ WVB (batched over 4 heads), ali = sigmoid(X3) @ W_ali ----
    gemm(X2bf, 0, DX2v, wVBbf, (long long)DX2v * 64, 64, 0,
         nullptr, zbbf, hQ, 64, nullptr, NROWS, 64, DX2v, 1.f, 0, 4);
    gemm(sX3bf, 0, DBv, wAlibf, 0, NROWS, 0,
         nullptr, alibf, 0, NROWS, nullptr, NROWS, NROWS, DBv, 1.f, 0, 1);

    // ---- per type-A head: scores -> column softmax -> S @ V into Zcat[:, h*64:] ----
    for (int h = 0; h < 8; ++h) {
        gemm(Qb + h * hQ, 0, 64, Kb + h * hQ, 0, 64, /*transB=*/1,
             scoresF, nullptr, 0, NROWS, nullptr, NROWS, NROWS, 64, 0.125f, 0, 1);
        col_softmax<<<dim3(NROWS / 256), dim3(256), 0, stream>>>(scoresF, Sbf, NROWS, NROWS);
        gemm(Sbf, 0, NROWS, Vb + h * hQ, 0, 64, 0,
             ZcatF + h * 64, nullptr, 0, 768, nullptr, NROWS, 64, NROWS, 1.f, 0, 1);
    }
    // ---- type-B heads: ZB = ali @ zb into Zcat[:, 512 + h*64] (batched, strideC=64) ----
    gemm(alibf, 0, NROWS, zbbf, hQ, 64, 0,
         ZcatF + MDL, nullptr, 64, 768, nullptr, NROWS, 64, NROWS, 1.f, 0, 4);

    // ---- Z = Zcat @ W_O ----
    cvt(ZcatF, Zcatbf, (size_t)NROWS * 768, 0);
    gemm(Zcatbf, 0, 768, wObf, 0, MDL, 0,
         xF, xbf, 0, MDL, nullptr, NROWS, MDL, 768, 1.f, 0, 1);

    // ---- two post-LN encoder layers (nhead=1) ----
    const float attScale = 0.04419417382415922f;   // 1/sqrt(512)
    for (int l = 0; l < 2; ++l) {
        const ushort_t* Wq = eWqbf + (size_t)l * MDL * MDL;
        const ushort_t* Wk = eWkbf + (size_t)l * MDL * MDL;
        const ushort_t* Wv = eWvbf + (size_t)l * MDL * MDL;
        const ushort_t* Wo = eWobf + (size_t)l * MDL * MDL;
        const ushort_t* W1 = eW1bf + (size_t)l * MDL * 64;
        const ushort_t* W2 = eW2bf + (size_t)l * 64 * MDL;

        gemm(xbf, 0, MDL, Wq, 0, MDL, 0, nullptr, qbf, 0, MDL, ebq + l * MDL, NROWS, MDL, MDL, 1.f, 0, 1);
        gemm(xbf, 0, MDL, Wk, 0, MDL, 0, nullptr, kbf, 0, MDL, ebk + l * MDL, NROWS, MDL, MDL, 1.f, 0, 1);
        gemm(xbf, 0, MDL, Wv, 0, MDL, 0, nullptr, vbf, 0, MDL, ebv + l * MDL, NROWS, MDL, MDL, 1.f, 0, 1);

        gemm(qbf, 0, MDL, kbf, 0, MDL, /*transB=*/1,
             scoresF, nullptr, 0, NROWS, nullptr, NROWS, NROWS, MDL, attScale, 0, 1);
        row_softmax<<<dim3(NROWS), dim3(256), 0, stream>>>(scoresF, Sbf, NROWS);
        gemm(Sbf, 0, NROWS, vbf, 0, MDL, 0,
             nullptr, attbf, 0, MDL, nullptr, NROWS, MDL, NROWS, 1.f, 0, 1);
        gemm(attbf, 0, MDL, Wo, 0, MDL, 0,
             projF, nullptr, 0, MDL, ebo + l * MDL, NROWS, MDL, MDL, 1.f, 0, 1);

        layernorm_res<<<dim3(NROWS), dim3(256), 0, stream>>>(
            xF, projF, g1p + l * MDL, b1p + l * MDL, x1F, x1bf, MDL);

        gemm(x1bf, 0, MDL, W1, 0, 64, 0,
             nullptr, h1bf, 0, 64, bW1p + l * 64, NROWS, 64, MDL, 1.f, /*relu=*/1, 1);
        gemm(h1bf, 0, 64, W2, 0, MDL, 0,
             fF, nullptr, 0, MDL, bW2p + l * MDL, NROWS, MDL, 64, 1.f, 0, 1);

        float*    outF = (l == 1) ? (float*)d_out : xF;
        ushort_t* outH = (l == 1) ? nullptr : xbf;
        layernorm_res<<<dim3(NROWS), dim3(256), 0, stream>>>(
            x1F, fF, g2p + l * MDL, b2p + l * MDL, outF, outH, MDL);
    }
}
